// SimplifiedDualChannelLSTM_62491774157147
// MI455X (gfx1250) — matrix operand order repaired and checked by reading, not compiled, not run on hardware
//
#include <hip/hip_runtime.h>

// ---------------------------------------------------------------------------
// MI455X (gfx1250) dual-channel LSTM inference — v3.
//  * Branch-free WMMA inner loops via compile-time tile counts (templates).
//  * B fragments batched per K-chunk: one load clause -> one wait -> NTILES
//    back-to-back v_wmma, instead of load/wait/wmma per tile.
//  * N padded so every wave owns an identical number of 16-wide N tiles;
//    padding lives as zeros in the pre-packed weights, so padded activation
//    columns contribute exactly zero (no buffer zero-init needed anywhere).
//  * Recurrence: 1 workgroup (8 waves) per 16 batch rows, all 128 timesteps;
//    h/c state + z scratch in LDS; packed U streamed from L2 (192 MB).
// ---------------------------------------------------------------------------

typedef __attribute__((ext_vector_type(16))) _Float16 v16h;
typedef __attribute__((ext_vector_type(8)))  _Float16 v8h;
typedef __attribute__((ext_vector_type(8)))  float    v8f;

#define DEVFN static __device__ __forceinline__

DEVFN float sigf(float x) { return 1.0f / (1.0f + __expf(-x)); }

// A fragment (16x32 f16): lane L -> row m=L&15; half=L>>4 selects K groups:
// f16[0..7] = K half*8 .. half*8+7 ; f16[8..15] = K 16+half*8 .. (ISA 7.12.2).
DEVFN v16h load_afrag(const _Float16* baseRow0, int lda, int lane, int kc) {
  const int m = lane & 15, half = lane >> 4;
  const _Float16* p = baseRow0 + (size_t)m * lda + kc * 32 + half * 8;
  v8h lo = *(const v8h*)p;
  v8h hi = *(const v8h*)(p + 16);
  v16h a;
#pragma unroll
  for (int i = 0; i < 8; ++i) { a[i] = lo[i]; a[i + 8] = hi[i]; }
  return a;
}

// B fragment (32x16 f16) from pre-packed weights: 16 contiguous f16 per lane.
DEVFN v16h load_bfrag(const _Float16* pk, int kc, int NTtot, int nt, int lane) {
  const _Float16* p = pk + ((((size_t)kc * NTtot + nt) * 32) + lane) * 16;
  v8h lo = *(const v8h*)p;
  v8h hi = *(const v8h*)(p + 8);
  v16h b;
#pragma unroll
  for (int i = 0; i < 8; ++i) { b[i] = lo[i]; b[i + 8] = hi[i]; }
  return b;
}

// ---------------------------------------------------------------------------
// Pack f32 row-major W[K x N] -> WMMA B-fragment order, f16, zero-padded
// (both K-pad rows and N-pad tiles are zeros -> pads never contribute).
// k = kc*32 + (val>>3)*16 + half*8 + (val&7) ; n = nt*16 + (lane&15). total = KC*NT*512.
// ---------------------------------------------------------------------------
__global__ void pack_b16(const float* __restrict__ W, _Float16* __restrict__ out,
                         int K, int N, int NT, int total) {
  int idx = blockIdx.x * 256 + threadIdx.x;
  if (idx >= total) return;
  int val = idx & 15;
  int lane = (idx >> 4) & 31;
  int tileIdx = idx >> 9;
  int nt = tileIdx % NT;
  int kc = tileIdx / NT;
  int j = val >> 1, e = val & 1;
  int half = lane >> 4, lanen = lane & 15;
  int k = kc * 32 + (val >> 3) * 16 + half * 8 + (val & 7);
  int n = nt * 16 + lanen;
  float f = (k < K && n < N) ? W[(size_t)k * N + n] : 0.0f;
  out[idx] = (_Float16)f;
}

// Precompute BN scale/shift: s = g*rsqrt(v+eps), t = b - m*s
__global__ void bn_prep(const float* __restrict__ g, const float* __restrict__ b,
                        const float* __restrict__ m, const float* __restrict__ v,
                        float* __restrict__ s, float* __restrict__ t, int n) {
  int i = blockIdx.x * 256 + threadIdx.x;
  if (i >= n) return;
  float sc = g[i] * rsqrtf(v[i] + 1e-3f);
  s[i] = sc;
  t[i] = b[i] - m[i] * sc;
}

// BN + f32 -> f16 conversion into padded row-major activation buffer.
__global__ void bn_apply_f16(const float* __restrict__ x, const float* __restrict__ s,
                             const float* __restrict__ t, _Float16* __restrict__ out,
                             int F, int ldo, long total) {
  long idx = (long)blockIdx.x * 256 + threadIdx.x;
  if (idx >= total) return;
  long r = idx / F;
  int f = (int)(idx - r * F);
  out[r * (size_t)ldo + f] = (_Float16)(x[idx] * s[f] + t[f]);
}

// ---------------------------------------------------------------------------
// Generic WMMA GEMM: out[M x n_out] = epi(A[M x Kpad] @ Bpk + bias)
//  block = 256 threads = 8 waves; wave owns 16 rows x NTILES N-tiles.
//  gridDim.y * NTILES == NTtot (exact; pad tiles are zero in packed B).
//  flags: 1 = remap output row r -> (r%T)*Bdim + r/T   ((B,T)->(T,B) layout)
//         2 = ReLU ;  postS/postT = optional post-ReLU affine (folded BN)
// ---------------------------------------------------------------------------
template <int NTILES>
__global__ void gemm16(const _Float16* __restrict__ A, int lda,
                       const _Float16* __restrict__ Bpk, int KC, int NTtot,
                       const float* __restrict__ bias,
                       const float* __restrict__ postS, const float* __restrict__ postT,
                       _Float16* __restrict__ out16, float* __restrict__ out32,
                       int ldo, int n_out, int Tdim, int Bdim, int flags) {
  const int lane = threadIdx.x & 31, wave = threadIdx.x >> 5;
  const int half = lane >> 4, lanen = lane & 15;
  const int r0 = (blockIdx.x * 8 + wave) * 16;
  const int nt0 = blockIdx.y * NTILES;

  v8f acc[NTILES];
#pragma unroll
  for (int j = 0; j < NTILES; ++j) {
    int n = (nt0 + j) * 16 + lanen;
    float bv = (bias && n < n_out) ? bias[n] : 0.0f;
#pragma unroll
    for (int i = 0; i < 8; ++i) acc[j][i] = bv;
  }

  const _Float16* Arow = A + (size_t)r0 * lda;
  for (int kc = 0; kc < KC; ++kc) {
    // batch the load clause, then issue all WMMAs back-to-back
    v16h a = load_afrag(Arow, lda, lane, kc);
    v16h bf[NTILES];
#pragma unroll
    for (int j = 0; j < NTILES; ++j)
      bf[j] = load_bfrag(Bpk, kc, NTtot, nt0 + j, lane);
#pragma unroll
    for (int j = 0; j < NTILES; ++j)
      acc[j] = __builtin_amdgcn_wmma_f32_16x16x32_f16(false, a, false, bf[j],
                                                      (short)0, acc[j], false, false);
  }

#pragma unroll
  for (int j = 0; j < NTILES; ++j) {
    int n = (nt0 + j) * 16 + lanen;
    if (n >= n_out) continue;
    float s = 1.0f, t = 0.0f;
    if (postS) { s = postS[n]; t = postT[n]; }
#pragma unroll
    for (int i = 0; i < 8; ++i) {
      int m = i + half * 8;
      long r = (long)r0 + m;
      long orow = (flags & 1) ? (r % Tdim) * (long)Bdim + (r / Tdim) : r;
      float v = acc[j][i];
      if (flags & 2) v = fmaxf(v, 0.0f);
      v = v * s + t;
      if (out16) out16[(size_t)orow * ldo + n] = (_Float16)v;
      else       out32[(size_t)orow * ldo + n] = v;
    }
  }
}

// ---------------------------------------------------------------------------
// LSTM recurrence: one workgroup = 16 batch rows, all T steps.
//  TPW = N tiles per wave (compile time). N4pad = TPW*128 >= 4H.
//  xz: precomputed x@W+b in (T, B, N4pad) f16 (pad cols: finite garbage that
//      meets zero-padded U rows/tiles -> contributes nothing & is never read
//      back by the gate stage).
//  Per step: z = xz_t + h@U (WMMA) -> LDS; gates on VALU; h/c updated in LDS;
//  h written out (optionally BN-folded via sOut/tOut) to (T, B, ldh) f16.
// ---------------------------------------------------------------------------
template <int TPW>
__global__ void lstm_rec(const _Float16* __restrict__ xz,
                         const _Float16* __restrict__ Upk,
                         _Float16* __restrict__ hsOut,
                         const float* __restrict__ sOut, const float* __restrict__ tOut,
                         int Bdim, int T, int H, int Hpad, int KC, int ldh) {
  constexpr int NT = TPW * 8;
  constexpr int N4 = NT * 16;
  __shared__ float    zbuf[16 * 640];   // 40 KB  (N4 <= 640)
  __shared__ _Float16 hbuf[16 * 160];   // 5 KB   (Hpad <= 160)
  __shared__ float    cbuf[16 * 160];   // 10 KB  (H <= 160)

  const int tid = threadIdx.x;
  const int lane = tid & 31, wave = tid >> 5;
  const int half = lane >> 4, lanen = lane & 15;
  const int b0 = blockIdx.x * 16;

  for (int i = tid; i < 16 * Hpad; i += 256) hbuf[i] = (_Float16)0.0f;
  for (int i = tid; i < 16 * H; i += 256) cbuf[i] = 0.0f;
  __syncthreads();

  for (int t = 0; t < T; ++t) {
    const size_t rowbase = (size_t)t * Bdim + b0;
    v8f acc[TPW];
#pragma unroll
    for (int j = 0; j < TPW; ++j) {
      int n = (wave * TPW + j) * 16 + lanen;
#pragma unroll
      for (int i = 0; i < 8; ++i) {
        int m = i + half * 8;
        acc[j][i] = (float)xz[(rowbase + m) * (size_t)N4 + n];
      }
    }
    for (int kc = 0; kc < KC; ++kc) {
      v16h a = load_afrag(hbuf, Hpad, lane, kc);
      v16h bf[TPW];
#pragma unroll
      for (int j = 0; j < TPW; ++j)
        bf[j] = load_bfrag(Upk, kc, NT, wave * TPW + j, lane);
#pragma unroll
      for (int j = 0; j < TPW; ++j)
        acc[j] = __builtin_amdgcn_wmma_f32_16x16x32_f16(false, a, false, bf[j],
                                                        (short)0, acc[j], false, false);
    }
#pragma unroll
    for (int j = 0; j < TPW; ++j) {
      int n = (wave * TPW + j) * 16 + lanen;
#pragma unroll
      for (int i = 0; i < 8; ++i) {
        int m = i + half * 8;
        zbuf[m * N4 + n] = acc[j][i];
      }
    }
    __syncthreads();

    for (int e = tid; e < 16 * H; e += 256) {
      int m = e / H, u = e - m * H;
      float zi = zbuf[m * N4 + u];
      float zf = zbuf[m * N4 + H + u];
      float zg = zbuf[m * N4 + 2 * H + u];
      float zo = zbuf[m * N4 + 3 * H + u];
      float c = cbuf[e];
      c = sigf(zf) * c + sigf(zi) * tanhf(zg);
      float h = sigf(zo) * tanhf(c);
      cbuf[e] = c;
      hbuf[m * Hpad + u] = (_Float16)h;
      float hv = h;
      if (sOut) hv = h * sOut[u] + tOut[u];
      hsOut[(rowbase + m) * (size_t)ldh + u] = (_Float16)hv;
    }
    __syncthreads();
  }
}

// ---------------------------------------------------------------------------
// Host launcher
// ---------------------------------------------------------------------------
extern "C" void kernel_launch(void* const* d_in, const int* in_sizes, int n_in,
                              void* d_out, int out_size, void* d_ws, size_t ws_size,
                              hipStream_t stream) {
  (void)in_sizes; (void)n_in; (void)out_size; (void)ws_size;
  const int B = 1024, T = 128, BT = B * T;

  const float* inputs = (const float*)d_in[0];
  const float* mbn1_g = (const float*)d_in[1];  const float* mbn1_b = (const float*)d_in[2];
  const float* mbn1_m = (const float*)d_in[3];  const float* mbn1_v = (const float*)d_in[4];
  const float* ml1_W  = (const float*)d_in[5];  const float* ml1_U  = (const float*)d_in[6];
  const float* ml1_bb = (const float*)d_in[7];
  const float* mbn2_g = (const float*)d_in[8];  const float* mbn2_b = (const float*)d_in[9];
  const float* mbn2_m = (const float*)d_in[10]; const float* mbn2_v = (const float*)d_in[11];
  const float* ml2_W  = (const float*)d_in[12]; const float* ml2_U  = (const float*)d_in[13];
  const float* ml2_bb = (const float*)d_in[14];
  const float* md_W   = (const float*)d_in[15]; const float* md_b   = (const float*)d_in[16];
  const float* abn1_g = (const float*)d_in[17]; const float* abn1_b = (const float*)d_in[18];
  const float* abn1_m = (const float*)d_in[19]; const float* abn1_v = (const float*)d_in[20];
  const float* al1_W  = (const float*)d_in[21]; const float* al1_U  = (const float*)d_in[22];
  const float* al1_bb = (const float*)d_in[23];
  const float* abn2_g = (const float*)d_in[24]; const float* abn2_b = (const float*)d_in[25];
  const float* abn2_m = (const float*)d_in[26]; const float* abn2_v = (const float*)d_in[27];
  const float* al2_W  = (const float*)d_in[28]; const float* al2_U  = (const float*)d_in[29];
  const float* al2_bb = (const float*)d_in[30];
  const float* ad_W   = (const float*)d_in[31]; const float* ad_b   = (const float*)d_in[32];
  const float* fbn_g  = (const float*)d_in[33]; const float* fbn_b  = (const float*)d_in[34];
  const float* fbn_m  = (const float*)d_in[35]; const float* fbn_v  = (const float*)d_in[36];
  const float* f1_W   = (const float*)d_in[37]; const float* f1_b   = (const float*)d_in[38];
  const float* f2_W   = (const float*)d_in[39]; const float* f2_b   = (const float*)d_in[40];
  const float* o_W    = (const float*)d_in[41]; const float* o_b    = (const float*)d_in[42];

  // ---- workspace carve-up ----
  char* ws = (char*)d_ws;
  size_t off = 0;
  auto carve = [&](size_t bytes) -> char* {
    char* p = ws + off;
    off = (off + bytes + 255) & ~(size_t)255;
    return p;
  };
  float* s_mbn1 = (float*)carve(119 * 4); float* t_mbn1 = (float*)carve(119 * 4);
  float* s_mbn2 = (float*)carve(160 * 4); float* t_mbn2 = (float*)carve(160 * 4);
  float* s_abn1 = (float*)carve(119 * 4); float* t_abn1 = (float*)carve(119 * 4);
  float* s_abn2 = (float*)carve(80 * 4);  float* t_abn2 = (float*)carve(80 * 4);
  float* s_fbn  = (float*)carve(144 * 4); float* t_fbn  = (float*)carve(144 * 4);

  auto pkbytes = [](int KC, int NT) { return (size_t)KC * NT * 512 * 2; };
  _Float16* pW1m = (_Float16*)carve(pkbytes(4, 40));  // 119x640
  _Float16* pU1m = (_Float16*)carve(pkbytes(5, 40));  // 160x640
  _Float16* pW2m = (_Float16*)carve(pkbytes(5, 24));  // 160x320 -> N padded 384
  _Float16* pU2m = (_Float16*)carve(pkbytes(3, 24));  //  80x320 -> 384
  _Float16* pW1a = (_Float16*)carve(pkbytes(4, 24));  // 119x320 -> 384
  _Float16* pU1a = (_Float16*)carve(pkbytes(3, 24));  //  80x320 -> 384
  _Float16* pW2a = (_Float16*)carve(pkbytes(3, 16));  //  80x160 -> 256
  _Float16* pU2a = (_Float16*)carve(pkbytes(2, 16));  //  40x160 -> 256
  _Float16* pMd  = (_Float16*)carve(pkbytes(3, 6));   //  80x96
  _Float16* pAd  = (_Float16*)carve(pkbytes(2, 3));   //  40x48
  _Float16* pF1  = (_Float16*)carve(pkbytes(5, 9));   // 144x144
  _Float16* pF2  = (_Float16*)carve(pkbytes(5, 5));   // 144x72 -> 80
  _Float16* pO   = (_Float16*)carve(pkbytes(3, 8));   //  72x119 -> 128

  _Float16* bnx   = (_Float16*)carve((size_t)BT * 128 * 2); // reused main/aux
  _Float16* xz1   = (_Float16*)carve((size_t)BT * 640 * 2); // reused main/aux
  _Float16* hs1   = (_Float16*)carve((size_t)BT * 160 * 2); // reused main/aux
  _Float16* xz2   = (_Float16*)carve((size_t)BT * 384 * 2); // reused main/aux
  _Float16* hs2m  = (_Float16*)carve((size_t)BT * 96 * 2);
  _Float16* hs2a  = (_Float16*)carve((size_t)BT * 64 * 2);
  _Float16* fused = (_Float16*)carve((size_t)B * 160 * 2);
  _Float16* buf1  = (_Float16*)carve((size_t)B * 160 * 2);
  _Float16* buf2  = (_Float16*)carve((size_t)B * 96 * 2);

  // ---- BN param prep ----
  bn_prep<<<1, 256, 0, stream>>>(mbn1_g, mbn1_b, mbn1_m, mbn1_v, s_mbn1, t_mbn1, 119);
  bn_prep<<<1, 256, 0, stream>>>(mbn2_g, mbn2_b, mbn2_m, mbn2_v, s_mbn2, t_mbn2, 160);
  bn_prep<<<1, 256, 0, stream>>>(abn1_g, abn1_b, abn1_m, abn1_v, s_abn1, t_abn1, 119);
  bn_prep<<<1, 256, 0, stream>>>(abn2_g, abn2_b, abn2_m, abn2_v, s_abn2, t_abn2, 80);
  bn_prep<<<1, 256, 0, stream>>>(fbn_g, fbn_b, fbn_m, fbn_v, s_fbn, t_fbn, 144);

  // ---- weight packing ----
  auto pack = [&](const float* W, _Float16* out, int K, int N, int NT, int KC) {
    int total = KC * NT * 512;
    pack_b16<<<(total + 255) / 256, 256, 0, stream>>>(W, out, K, N, NT, total);
  };
  pack(ml1_W, pW1m, 119, 640, 40, 4);
  pack(ml1_U, pU1m, 160, 640, 40, 5);
  pack(ml2_W, pW2m, 160, 320, 24, 5);
  pack(ml2_U, pU2m, 80, 320, 24, 3);
  pack(al1_W, pW1a, 119, 320, 24, 4);
  pack(al1_U, pU1a, 80, 320, 24, 3);
  pack(al2_W, pW2a, 80, 160, 16, 3);
  pack(al2_U, pU2a, 40, 160, 16, 2);
  pack(md_W, pMd, 80, 96, 6, 3);
  pack(ad_W, pAd, 40, 48, 3, 2);
  pack(f1_W, pF1, 144, 144, 9, 5);
  pack(f2_W, pF2, 144, 72, 5, 5);
  pack(o_W, pO, 72, 119, 8, 3);

  const long totIn = (long)BT * 119;
  const int bnBlocks = (int)((totIn + 255) / 256);

  // ================= main channel =================
  bn_apply_f16<<<bnBlocks, 256, 0, stream>>>(inputs, s_mbn1, t_mbn1, bnx, 119, 128, totIn);
  // xz1 = bn(x) @ W1 + b, written (T,B,640) (flags=1: remap rows)
  gemm16<4><<<dim3(BT / 128, 10), 256, 0, stream>>>(bnx, 128, pW1m, 4, 40, ml1_bb,
      nullptr, nullptr, xz1, nullptr, 640, 640, T, B, 1);
  // LSTM1 (H=160, N4=640 -> TPW=5), BN(mbn2) folded into h output
  lstm_rec<5><<<B / 16, 256, 0, stream>>>(xz1, pU1m, hs1, s_mbn2, t_mbn2, B, T, 160, 160, 5, 160);
  // xz2 = bn(h1) @ W2 + b, (T,B,384 ld), 320 real cols
  gemm16<4><<<dim3(BT / 128, 6), 256, 0, stream>>>(hs1, 160, pW2m, 5, 24, ml2_bb,
      nullptr, nullptr, xz2, nullptr, 384, 320, 1, B, 0);
  // LSTM2 (H=80, N4pad=384 -> TPW=3)
  lstm_rec<3><<<B / 16, 256, 0, stream>>>(xz2, pU2m, hs2m, nullptr, nullptr, B, T, 80, 96, 3, 96);

  // ================= aux channel =================
  bn_apply_f16<<<bnBlocks, 256, 0, stream>>>(inputs, s_abn1, t_abn1, bnx, 119, 128, totIn);
  gemm16<4><<<dim3(BT / 128, 6), 256, 0, stream>>>(bnx, 128, pW1a, 4, 24, al1_bb,
      nullptr, nullptr, xz1, nullptr, 384, 320, T, B, 1);
  lstm_rec<3><<<B / 16, 256, 0, stream>>>(xz1, pU1a, hs1, s_abn2, t_abn2, B, T, 80, 96, 3, 96);
  gemm16<4><<<dim3(BT / 128, 4), 256, 0, stream>>>(hs1, 96, pW2a, 3, 16, al2_bb,
      nullptr, nullptr, xz2, nullptr, 256, 160, 1, B, 0);
  lstm_rec<2><<<B / 16, 256, 0, stream>>>(xz2, pU2a, hs2a, nullptr, nullptr, B, T, 40, 64, 2, 64);

  // ================= fusion head =================
  const _Float16* h2m = hs2m + (size_t)(T - 1) * B * 96;
  const _Float16* h2a = hs2a + (size_t)(T - 1) * B * 64;
  // mx = fbn(relu(h2m @ md_W + b)) -> fused[:, 0:96]
  gemm16<3><<<dim3(B / 128, 2), 256, 0, stream>>>(h2m, 96, pMd, 3, 6, md_b,
      s_fbn, t_fbn, fused, nullptr, 160, 96, 1, B, 2);
  // ax = fbn(relu(h2a @ ad_W + b)) -> fused[:, 96:144]
  gemm16<3><<<dim3(B / 128, 1), 256, 0, stream>>>(h2a, 64, pAd, 2, 3, ad_b,
      s_fbn + 96, t_fbn + 96, fused + 96, nullptr, 160, 48, 1, B, 2);
  // f1: relu(fused @ f1_W + b)   (K=144 padded to 160; packed zeros cover pad)
  gemm16<3><<<dim3(B / 128, 3), 256, 0, stream>>>(fused, 160, pF1, 5, 9, f1_b,
      nullptr, nullptr, buf1, nullptr, 160, 144, 1, B, 2);
  // f2: relu(buf1 @ f2_W + b)
  gemm16<5><<<dim3(B / 128, 1), 256, 0, stream>>>(buf1, 160, pF2, 5, 5, f2_b,
      nullptr, nullptr, buf2, nullptr, 96, 72, 1, B, 2);
  // out: buf2 @ o_W + b -> f32 d_out (1024 x 119)
  gemm16<4><<<dim3(B / 128, 2), 256, 0, stream>>>(buf2, 96, pO, 3, 8, o_b,
      nullptr, nullptr, nullptr, (float*)d_out, 119, 119, 1, B, 0);
}
